// MultiHeadAttention_40097814676020
// MI455X (gfx1250) — compile-verified, matrix operand order checked by reading, not run on hardware
//
#include <hip/hip_runtime.h>
#include <hip/hip_bf16.h>

#define D_MODEL 1024
#define L_SEQ   1024
#define NHEADS  16
#define HDIM    64
#define BATCH   4
#define LN_EPS  1e-6f

typedef __attribute__((ext_vector_type(16))) __bf16       v16bf;
typedef __attribute__((ext_vector_type(2)))  __bf16       v2bf;
typedef __attribute__((ext_vector_type(8)))  float        v8f;
typedef __attribute__((ext_vector_type(8)))  unsigned int v8u;

union Frag {
    v8u   u;
    v16bf b;
};

// f32 -> bf16 via the first-class __bf16 type: lets ISel pick the native
// gfx1250 convert (packed where possible) instead of a manual bit sequence.
__device__ __forceinline__ unsigned short f2bf_u16(float f) {
    __bf16 h = (__bf16)f;
    unsigned short u;
    __builtin_memcpy(&u, &h, 2);
    return u;
}
__device__ __forceinline__ unsigned int pack2(float lo, float hi) {
    v2bf t;
    t[0] = (__bf16)lo;
    t[1] = (__bf16)hi;
    unsigned int u;
    __builtin_memcpy(&u, &t, 4);
    return u;
}

__device__ __forceinline__ v8u combine8(uint4 a, uint4 b) {
    v8u r;
    r[0] = a.x; r[1] = a.y; r[2] = a.z; r[3] = a.w;
    r[4] = b.x; r[5] = b.y; r[6] = b.z; r[7] = b.w;
    return r;
}
__device__ __forceinline__ v8f zero8() {
    v8f z;
#pragma unroll
    for (int i = 0; i < 8; ++i) z[i] = 0.0f;
    return z;
}
__device__ __forceinline__ v8f wmma_bf16(Frag a, Frag b, v8f c) {
    // (neg_a, A, neg_b, B, c_mod, C, reuse_a, reuse_b)
    return __builtin_amdgcn_wmma_f32_16x16x32_bf16(false, a.b, false, b.b,
                                                   (short)0, c, false, false);
}

// ---------------------------------------------------------------------------
// Kernel 1: P = X @ W^T + b  (grid.z selects q/k/v).
//   mode 0: qp -> f32 [B*L,D] and bf16 [B*L,D]
//   mode 1: kp -> bf16 [B*L,D]
//   mode 2: vp -> bf16 transposed [B, D, L]   (contiguous-K B-operand for ctx GEMM)
// Block: 256 thr (8 waves, 4x2), tile 128x64, BK=32, LDS-staged bf16.
// ---------------------------------------------------------------------------
__global__ __launch_bounds__(256)
void qkv_proj_kernel(const float* __restrict__ q,  const float* __restrict__ k,
                     const float* __restrict__ v,
                     const float* __restrict__ Wq, const float* __restrict__ Wk,
                     const float* __restrict__ Wv,
                     const float* __restrict__ bq, const float* __restrict__ bk,
                     const float* __restrict__ bv,
                     float* __restrict__ qpF,
                     unsigned short* __restrict__ qpB,
                     unsigned short* __restrict__ kpB,
                     unsigned short* __restrict__ vT)
{
    const int mode = blockIdx.z;
    const float* X; const float* W; const float* bias;
    if (mode == 0)      { X = q; W = Wq; bias = bq; }
    else if (mode == 1) { X = k; W = Wk; bias = bk; }
    else                { X = v; W = Wv; bias = bv; }

    const int tid  = threadIdx.x;
    const int wave = tid >> 5;
    const int lane = tid & 31;
    const int wm   = wave >> 1;      // 0..3
    const int wn   = wave & 1;       // 0..1
    const int hi16 = lane >> 4;      // lane half
    const int l15  = lane & 15;

    const int m0 = blockIdx.x * 128;
    const int n0 = blockIdx.y * 64;

    __shared__ __align__(16) unsigned int As[128 * 16]; // 128 rows x 32 K (bf16 pairs)
    __shared__ __align__(16) unsigned int Bs[64 * 16];  // 64 rows x 32 K

    v8f acc[2][2];
#pragma unroll
    for (int i = 0; i < 2; ++i)
#pragma unroll
        for (int j = 0; j < 2; ++j) acc[i][j] = zero8();

    for (int k0 = 0; k0 < D_MODEL; k0 += 32) {
        { // stage A tile: 256 thr, 16 floats each -> bf16 pack -> LDS
            const int row = tid >> 1;
            const int kk  = (tid & 1) * 16;
            const float* src = X + (size_t)(m0 + row) * D_MODEL + k0 + kk;
            if (k0 + 32 < D_MODEL) __builtin_prefetch(src + 32, 0, 1); // global_prefetch
            float4 f0 = *(const float4*)(src);
            float4 f1 = *(const float4*)(src + 4);
            float4 f2 = *(const float4*)(src + 8);
            float4 f3 = *(const float4*)(src + 12);
            uint4 p0 = make_uint4(pack2(f0.x, f0.y), pack2(f0.z, f0.w),
                                  pack2(f1.x, f1.y), pack2(f1.z, f1.w));
            uint4 p1 = make_uint4(pack2(f2.x, f2.y), pack2(f2.z, f2.w),
                                  pack2(f3.x, f3.y), pack2(f3.z, f3.w));
            *(uint4*)&As[row * 16 + kk / 2]     = p0;
            *(uint4*)&As[row * 16 + kk / 2 + 4] = p1;
        }
        { // stage B tile: 256 thr, 8 floats each
            const int row = tid >> 2;
            const int kk  = (tid & 3) * 8;
            const float* src = W + (size_t)(n0 + row) * D_MODEL + k0 + kk;
            float4 f0 = *(const float4*)(src);
            float4 f1 = *(const float4*)(src + 4);
            uint4 p = make_uint4(pack2(f0.x, f0.y), pack2(f0.z, f0.w),
                                 pack2(f1.x, f1.y), pack2(f1.z, f1.w));
            *(uint4*)&Bs[row * 16 + kk / 2] = p;
        }
        __syncthreads();

        Frag fa[2], fb[2];
#pragma unroll
        for (int i = 0; i < 2; ++i) {
            const int r = wm * 32 + i * 16 + l15;
            uint4 c0 = *(const uint4*)&As[r * 16 + hi16 * 4];     // K {0..7}/{8..15}
            uint4 c1 = *(const uint4*)&As[r * 16 + 8 + hi16 * 4]; // K {16..23}/{24..31}
            fa[i].u = combine8(c0, c1);
        }
#pragma unroll
        for (int j = 0; j < 2; ++j) {
            const int r = wn * 32 + j * 16 + l15;
            uint4 c0 = *(const uint4*)&Bs[r * 16 + hi16 * 4];
            uint4 c1 = *(const uint4*)&Bs[r * 16 + 8 + hi16 * 4];
            fb[j].u = combine8(c0, c1);
        }
#pragma unroll
        for (int i = 0; i < 2; ++i)
#pragma unroll
            for (int j = 0; j < 2; ++j)
                acc[i][j] = wmma_bf16(fa[i], fb[j], acc[i][j]);
        __syncthreads();
    }

    // epilogue: C layout VGPR e -> M = e + hi16*8, N = l15
#pragma unroll
    for (int i = 0; i < 2; ++i) {
#pragma unroll
        for (int j = 0; j < 2; ++j) {
            const int   nco  = n0 + wn * 32 + j * 16 + l15;
            const float badd = bias[nco];
#pragma unroll
            for (int e = 0; e < 8; ++e) {
                const int   mro = m0 + wm * 32 + i * 16 + hi16 * 8 + e;
                const float val = acc[i][j][e] + badd;
                if (mode == 0) {
                    qpF[(size_t)mro * D_MODEL + nco] = val;
                    qpB[(size_t)mro * D_MODEL + nco] = f2bf_u16(val);
                } else if (mode == 1) {
                    kpB[(size_t)mro * D_MODEL + nco] = f2bf_u16(val);
                } else {
                    const int bb = mro >> 10;        // batch
                    const int ll = mro & 1023;       // seq pos
                    vT[((size_t)bb * D_MODEL + nco) * L_SEQ + ll] = f2bf_u16(val);
                }
            }
        }
    }
}

// ---------------------------------------------------------------------------
// Kernel 2: scores = (Qh Kh^T)/8, mask, softmax -> attention probs (f32).
// Grid (L/16, H, B); block = 8 waves; wave w owns score columns [128w,128w+128).
// Scores tile (16x1024 f32 = 64KB) stays in LDS; probs written once to HBM.
// ---------------------------------------------------------------------------
__global__ __launch_bounds__(256)
void attn_softmax_kernel(const unsigned short* __restrict__ qpB,
                         const unsigned short* __restrict__ kpB,
                         const unsigned char*  __restrict__ masks,
                         float* __restrict__ attn)
{
    const int b    = blockIdx.z;
    const int h    = blockIdx.y;
    const int q0   = blockIdx.x * 16;
    const int tid  = threadIdx.x;
    const int wave = tid >> 5;
    const int lane = tid & 31;
    const int hi16 = lane >> 4;
    const int l15  = lane & 15;

    __shared__ __align__(16) float S[16 * L_SEQ];

    // A fragments: Q rows q0..q0+15, K = Dh = 64 -> two 16x32 bf16 frags
    const unsigned short* qrow =
        qpB + (size_t)(b * L_SEQ + q0 + l15) * D_MODEL + h * HDIM;
    Frag fa[2];
#pragma unroll
    for (int s = 0; s < 2; ++s) {
        uint4 c0 = *(const uint4*)(qrow + s * 32 + hi16 * 8);
        uint4 c1 = *(const uint4*)(qrow + s * 32 + 16 + hi16 * 8);
        fa[s].u = combine8(c0, c1);
    }

#pragma unroll
    for (int t = 0; t < 8; ++t) {
        const int j0 = wave * 128 + t * 16;
        const unsigned short* krow =
            kpB + (size_t)(b * L_SEQ + j0 + l15) * D_MODEL + h * HDIM;
        v8f acc = zero8();
#pragma unroll
        for (int s = 0; s < 2; ++s) {
            uint4 c0 = *(const uint4*)(krow + s * 32 + hi16 * 8);
            uint4 c1 = *(const uint4*)(krow + s * 32 + 16 + hi16 * 8);
            Frag fb; fb.u = combine8(c0, c1);
            acc = wmma_bf16(fa[s], fb, acc);
        }
#pragma unroll
        for (int e = 0; e < 8; ++e) {
            const int r = hi16 * 8 + e;
            S[r * L_SEQ + j0 + l15] = acc[e] * 0.125f;   // 1/sqrt(64)
        }
    }
    __syncthreads();

    // softmax: 2 rows per wave, 32 cols per lane, wave32 shuffle reductions
    for (int rr = 0; rr < 2; ++rr) {
        const int r  = wave * 2 + rr;
        const int qg = q0 + r;
        const unsigned char* mrow = masks + ((size_t)b * L_SEQ + qg) * L_SEQ;
        float vals[32];
        float mx = -3.0e38f;
#pragma unroll
        for (int i = 0; i < 32; ++i) {
            const int c = i * 32 + lane;
            float s = S[r * L_SEQ + c];
            if (mrow[c]) s = -__builtin_inff();
            vals[i] = s;
            mx = fmaxf(mx, s);
        }
#pragma unroll
        for (int off = 16; off > 0; off >>= 1) mx = fmaxf(mx, __shfl_xor(mx, off, 32));
        float sum = 0.0f;
#pragma unroll
        for (int i = 0; i < 32; ++i) { float e = __expf(vals[i] - mx); vals[i] = e; sum += e; }
#pragma unroll
        for (int off = 16; off > 0; off >>= 1) sum += __shfl_xor(sum, off, 32);
        const float inv = 1.0f / sum;
        float* arow = attn + (((size_t)(b * NHEADS + h)) * L_SEQ + qg) * L_SEQ;
#pragma unroll
        for (int i = 0; i < 32; ++i) arow[i * 32 + lane] = vals[i] * inv;
    }
}

// ---------------------------------------------------------------------------
// Kernel 3: ctx = attn @ V (per head), concat heads, +qp residual, LayerNorm.
// Grid (L/16, B); block = 8 waves; wave owns 2 heads x 4 n-tiles (A-frag reused
// across the 4 WMMAs). ctx tile 16x1024 f32 in LDS, then LN per row.
// ---------------------------------------------------------------------------
__global__ __launch_bounds__(256)
void ctx_ln_kernel(const float* __restrict__ attn,
                   const unsigned short* __restrict__ vT,
                   const float* __restrict__ qpF,
                   const float* __restrict__ ln_g,
                   const float* __restrict__ ln_b,
                   float* __restrict__ out)
{
    const int b    = blockIdx.y;
    const int q0   = blockIdx.x * 16;
    const int tid  = threadIdx.x;
    const int wave = tid >> 5;
    const int lane = tid & 31;
    const int hi16 = lane >> 4;
    const int l15  = lane & 15;

    __shared__ __align__(16) float Cs[16 * D_MODEL];

    for (int hh = 0; hh < 2; ++hh) {
        const int h = wave * 2 + hh;
        const float* arow =
            attn + (((size_t)(b * NHEADS + h)) * L_SEQ + q0 + l15) * L_SEQ;
        const unsigned short* vbase =
            vT + ((size_t)b * D_MODEL + h * HDIM + l15) * L_SEQ;
        v8f acc[4];
#pragma unroll
        for (int j = 0; j < 4; ++j) acc[j] = zero8();

        for (int k0 = 0; k0 < L_SEQ; k0 += 32) {
            // A frag: attention row (f32) -> bf16 pack
            const float* a0 = arow + k0 + hi16 * 8;
            const float* a1 = arow + k0 + 16 + hi16 * 8;
            float4 f0 = *(const float4*)(a0);
            float4 f1 = *(const float4*)(a0 + 4);
            float4 f2 = *(const float4*)(a1);
            float4 f3 = *(const float4*)(a1 + 4);
            Frag fa;
            fa.u[0] = pack2(f0.x, f0.y); fa.u[1] = pack2(f0.z, f0.w);
            fa.u[2] = pack2(f1.x, f1.y); fa.u[3] = pack2(f1.z, f1.w);
            fa.u[4] = pack2(f2.x, f2.y); fa.u[5] = pack2(f2.z, f2.w);
            fa.u[6] = pack2(f3.x, f3.y); fa.u[7] = pack2(f3.z, f3.w);
#pragma unroll
            for (int j = 0; j < 4; ++j) {
                const unsigned short* vrow = vbase + (size_t)j * 16 * L_SEQ;
                uint4 c0 = *(const uint4*)(vrow + k0 + hi16 * 8);
                uint4 c1 = *(const uint4*)(vrow + k0 + 16 + hi16 * 8);
                Frag fb; fb.u = combine8(c0, c1);
                acc[j] = wmma_bf16(fa, fb, acc[j]);
            }
        }
#pragma unroll
        for (int j = 0; j < 4; ++j)
#pragma unroll
            for (int e = 0; e < 8; ++e) {
                const int r = hi16 * 8 + e;
                Cs[r * D_MODEL + h * HDIM + j * 16 + l15] = acc[j][e];
            }
    }
    __syncthreads();

    // residual + LayerNorm: 2 rows per wave
    for (int rr = 0; rr < 2; ++rr) {
        const int    r       = wave * 2 + rr;
        const size_t rowbase = ((size_t)(b * L_SEQ + q0 + r)) * D_MODEL;
        float sum = 0.0f, sum2 = 0.0f;
        for (int c = lane; c < D_MODEL; c += 32) {
            const float x = qpF[rowbase + c] + Cs[r * D_MODEL + c];
            sum += x; sum2 += x * x;
        }
#pragma unroll
        for (int off = 16; off > 0; off >>= 1) {
            sum  += __shfl_xor(sum,  off, 32);
            sum2 += __shfl_xor(sum2, off, 32);
        }
        const float mu  = sum  * (1.0f / D_MODEL);
        const float var = sum2 * (1.0f / D_MODEL) - mu * mu;
        const float rs  = rsqrtf(var + LN_EPS);
        for (int c = lane; c < D_MODEL; c += 32) {
            const float x = qpF[rowbase + c] + Cs[r * D_MODEL + c];
            out[rowbase + c] = (x - mu) * rs * ln_g[c] + ln_b[c];
        }
    }
}

// ---------------------------------------------------------------------------
extern "C" void kernel_launch(void* const* d_in, const int* in_sizes, int n_in,
                              void* d_out, int out_size, void* d_ws, size_t ws_size,
                              hipStream_t stream)
{
    (void)in_sizes; (void)n_in; (void)out_size; (void)ws_size;

    const float* q  = (const float*)d_in[0];
    const float* k  = (const float*)d_in[1];
    const float* v  = (const float*)d_in[2];
    const unsigned char* masks = (const unsigned char*)d_in[3];
    const float* Wq = (const float*)d_in[4];
    const float* bq = (const float*)d_in[5];
    const float* Wk = (const float*)d_in[6];
    const float* bk = (const float*)d_in[7];
    const float* Wv = (const float*)d_in[8];
    const float* bv = (const float*)d_in[9];
    const float* ln_g = (const float*)d_in[10];
    const float* ln_b = (const float*)d_in[11];

    float* out  = (float*)d_out;                                // [4,1024,1024]
    float* attn = out + (size_t)BATCH * L_SEQ * D_MODEL;        // [64,1024,1024]

    char* w = (char*)d_ws;
    float* qpF = (float*)w;                    w += (size_t)BATCH * L_SEQ * D_MODEL * sizeof(float);
    unsigned short* qpB = (unsigned short*)w;  w += (size_t)BATCH * L_SEQ * D_MODEL * sizeof(unsigned short);
    unsigned short* kpB = (unsigned short*)w;  w += (size_t)BATCH * L_SEQ * D_MODEL * sizeof(unsigned short);
    unsigned short* vT  = (unsigned short*)w;  // bf16 [B, D, L]

    dim3 grid1(BATCH * L_SEQ / 128, D_MODEL / 64, 3);
    qkv_proj_kernel<<<grid1, dim3(256), 0, stream>>>(q, k, v, Wq, Wk, Wv,
                                                     bq, bk, bv,
                                                     qpF, qpB, kpB, vT);

    dim3 grid2(L_SEQ / 16, NHEADS, BATCH);
    attn_softmax_kernel<<<grid2, dim3(256), 0, stream>>>(qpB, kpB, masks, attn);

    dim3 grid3(L_SEQ / 16, BATCH);
    ctx_ln_kernel<<<grid3, dim3(256), 0, stream>>>(attn, vT, qpF, ln_g, ln_b, out);
}
